// MonotonicAttention_35192962023606
// MI455X (gfx1250) — matrix operand order repaired and checked by reading, not compiled, not run on hardware
//
#include <hip/hip_runtime.h>

// MonotonicAttention fused kernel for gfx1250 (MI455X), wave32.
// One wave (32 lanes) per (batch b, 16-query tile qt).
//   Pass 1: W = Q K^T / sqrt(64), causal mask, tiles via v_wmma_f32_16x16x32_bf16 -> LDS fp32
//   Pass 2: per-row softmax -> lane-scan cumsum -> distance reweight -> second softmax -> S in LDS
//   Pass 3: C = S V via v_wmma_f32_16x16x32_bf16, fp32 output
// All score traffic stays in LDS (16 x 1025 fp32 = 65,600 B, row-padded for bank-conflict-free
// column gathers); HBM traffic is just Q,K,V,out (~32 MB total across the dispatch).

typedef __attribute__((ext_vector_type(16))) __bf16 v16bf;
typedef __attribute__((ext_vector_type(8)))  float  v8f;

#define NQ 1024
#define ND 64
#define LSTR 1025                      // LDS row stride in floats (pad 1 dword)
#define NEG_W (-12500000.0f)           // NEG_INF / sqrt(dk) = -1e8 / 8

__global__ __launch_bounds__(32)
void monotonic_attn_kernel(const float* __restrict__ Q,
                           const float* __restrict__ K,
                           const float* __restrict__ V,
                           const float* __restrict__ gamma,
                           float* __restrict__ out)
{
    extern __shared__ float lds[];     // 16 rows x LSTR floats

    const int lane = threadIdx.x;      // 0..31
    const int half = lane >> 4;        // half-wave select
    const int lm   = lane & 15;
    const int qt   = blockIdx.x;       // query tile 0..63
    const int b    = blockIdx.y;       // batch 0..31
    const int i0   = qt * 16;
    const int ntile = qt + 1;          // causal key tiles

    // theta = -softplus(gamma), scalar
    const float theta = -__logf(1.0f + __expf(gamma[0]));

    // ---------------- Q A-fragments: two 16x32 bf16 chunks over D=64 ----------------
    // A operand as packed here (16x32): lane gives M=l%16; elem j -> K = j + 16*half (same k order as B)
    const float* qrow = Q + ((size_t)b * NQ + (size_t)(i0 + lm)) * ND;
    v16bf aQ0, aQ1;
    #pragma unroll
    for (int j = 0; j < 16; ++j) {
        const int kk = j + 16 * half;
        aQ0[j] = (__bf16)qrow[kk];
        aQ1[j] = (__bf16)qrow[32 + kk];
    }

    // ---------------- Pass 1: masked scaled scores into LDS ----------------
    for (int t = 0; t < ntile; ++t) {
        const int m0 = t * 16;
        const float* krow = K + ((size_t)b * NQ + (size_t)(m0 + lm)) * ND;
        if (t + 1 < ntile)  // global_prefetch_b8 for next K tile
            __builtin_prefetch(K + ((size_t)b * NQ + (size_t)(m0 + 16 + lm)) * ND, 0, 3);

        // B layout (32x16): lane gives N=key=l%16; elem j -> K(dim) = j + 16*half
        v16bf bK0, bK1;
        #pragma unroll
        for (int j = 0; j < 16; ++j) {
            bK0[j] = (__bf16)krow[j + 16 * half];
            bK1[j] = (__bf16)krow[32 + j + 16 * half];
        }

        v8f acc = {};
        acc = __builtin_amdgcn_wmma_f32_16x16x32_bf16(false, aQ0, false, bK0,
                                                      (short)0, acc, false, false);
        acc = __builtin_amdgcn_wmma_f32_16x16x32_bf16(false, aQ1, false, bK1,
                                                      (short)0, acc, false, false);

        // C layout: lane N=l%16 (key), vgpr v -> M = v + 8*half (query)
        #pragma unroll
        for (int v = 0; v < 8; ++v) {
            const int row = v + 8 * half;
            const int m   = m0 + lm;
            float w = acc[v] * 0.125f;               // / sqrt(64)
            if (m > i0 + row) w = NEG_W;             // causal mask (mask input == triu)
            lds[row * LSTR + m] = w;
        }
    }
    __syncthreads();

    // ---------------- Pass 2: softmax -> cumsum -> reweight -> softmax ----------------
    const int K2 = (ntile * 16 + 31) & ~31;          // keys padded to wmma K=32 boundary
    for (int r = 0; r < 16; ++r) {
        const int i   = i0 + r;
        const int len = i + 1;                        // valid causal length
        float* wrow = lds + r * LSTR;

        // row max
        float mx = -3.0e38f;
        for (int c = lane; c < len; c += 32) mx = fmaxf(mx, wrow[c]);
        #pragma unroll
        for (int off = 16; off; off >>= 1) mx = fmaxf(mx, __shfl_xor(mx, off, 32));

        // sum of exp
        float sm = 0.0f;
        for (int c = lane; c < len; c += 32) sm += __expf(wrow[c] - mx);
        #pragma unroll
        for (int off = 16; off; off >>= 1) sm += __shfl_xor(sm, off, 32);
        const float inv = 1.0f / sm;

        // inclusive scan of score; disttotal == 1 for softmax, so rem = 1 - prefix.
        // Fused: w2 = w * clip(exp(sqrt(rem*(i-c)) * theta), 1e-5, 1e5); track max2.
        float run = 0.0f;
        float mx2 = -3.0e38f;
        for (int c0 = 0; c0 < len; c0 += 32) {
            const int c = c0 + lane;
            const float w = (c < len) ? wrow[c] : 0.0f;
            float x = (c < len) ? __expf(w - mx) * inv : 0.0f;
            #pragma unroll
            for (int off = 1; off < 32; off <<= 1) {
                const float t = __shfl_up(x, off, 32);
                if (lane >= off) x += t;
            }
            const float pre = run + x;               // inclusive prefix (distcum)
            run += __shfl(x, 31, 32);                // carry chunk total
            if (c < len) {
                const float rem  = fmaxf(1.0f - pre, 0.0f);
                const float dist = sqrtf(rem * (float)(i - c));
                float te = __expf(dist * theta);
                te = fminf(fmaxf(te, 1e-5f), 1e5f);
                const float w2 = w * te;
                wrow[c] = w2;
                mx2 = fmaxf(mx2, w2);
            }
        }
        #pragma unroll
        for (int off = 16; off; off >>= 1) mx2 = fmaxf(mx2, __shfl_xor(mx2, off, 32));

        float sm2 = 0.0f;
        for (int c = lane; c < len; c += 32) sm2 += __expf(wrow[c] - mx2);
        #pragma unroll
        for (int off = 16; off; off >>= 1) sm2 += __shfl_xor(sm2, off, 32);
        const float inv2 = 1.0f / sm2;

        // final attention weights S, zero-padded to K2 for full-tile GEMM2
        for (int c = lane; c < K2; c += 32)
            wrow[c] = (c < len) ? __expf(wrow[c] - mx2) * inv2 : 0.0f;
    }
    __syncthreads();

    // ---------------- Pass 3: C = S V ----------------
    v8f accO[4] = {};                                 // 4 d-tiles of 16 cols
    for (int kb = 0; kb < K2; kb += 32) {
        // A = S[16 x 32] gathered from LDS (row stride 1025 => conflict-free columns)
        v16bf aS;
        #pragma unroll
        for (int j = 0; j < 16; ++j) {
            const int kk = j + 16 * half;
            aS[j] = (__bf16)lds[lm * LSTR + kb + kk];
        }
        #pragma unroll
        for (int dt = 0; dt < 4; ++dt) {
            // B = V[32 keys x 16 dims]: lane N = d = dt*16 + l%16; elem j -> key = kb + j + 16*half
            v16bf bV;
            #pragma unroll
            for (int j = 0; j < 16; ++j) {
                const int key = kb + j + 16 * half;
                bV[j] = (__bf16)V[((size_t)b * NQ + (size_t)key) * ND + dt * 16 + lm];
            }
            accO[dt] = __builtin_amdgcn_wmma_f32_16x16x32_bf16(false, aS, false, bV,
                                                               (short)0, accO[dt], false, false);
        }
    }

    // store 16x64 fp32 output tile
    #pragma unroll
    for (int dt = 0; dt < 4; ++dt) {
        #pragma unroll
        for (int v = 0; v < 8; ++v) {
            const int row = v + 8 * half;
            out[((size_t)b * NQ + (size_t)(i0 + row)) * ND + dt * 16 + lm] = accO[dt][v];
        }
    }
}

extern "C" void kernel_launch(void* const* d_in, const int* in_sizes, int n_in,
                              void* d_out, int out_size, void* d_ws, size_t ws_size,
                              hipStream_t stream) {
    (void)in_sizes; (void)n_in; (void)out_size; (void)d_ws; (void)ws_size;
    const float* Q     = (const float*)d_in[0];
    const float* K     = (const float*)d_in[1];
    const float* V     = (const float*)d_in[2];
    // d_in[3] is the causal mask (bool) -- applied analytically in-kernel.
    const float* gamma = (const float*)d_in[4];
    // d_in[5] is dk == 64, baked into the 0.125f scale.
    float* out = (float*)d_out;

    dim3 grid(NQ / 16, 32);            // 64 query tiles x 32 batches
    dim3 block(32);                    // one wave32 per tile
    const size_t shmem = (size_t)16 * LSTR * sizeof(float);   // 65,600 B of LDS
    monotonic_attn_kernel<<<grid, block, shmem, stream>>>(Q, K, V, gamma, out);
}
